// snn_model_28793460752746
// MI455X (gfx1250) — compile-verified, matrix operand order checked by reading, not run on hardware
//
#include <hip/hip_runtime.h>
#include <stdint.h>

// ---------------------------------------------------------------------------
// SNN (LIF) network on MI455X / gfx1250.
// All matmuls/convs run as fp16 WMMA (v_wmma_f32_16x16x32_f16) with f32
// accumulate. Spikes are exactly representable in fp16 ({0,1}); weights are
// converted fp32->fp16 once per launch (Wl1 drops from 302MB fp32 -> 151MB
// fp16, fitting the 192MB L2 so the dominant lin1 GEMM runs at L2 bandwidth).
// LIF update is fused into each GEMM epilogue; membrane state is f32 in ws.
// ---------------------------------------------------------------------------

typedef __attribute__((ext_vector_type(16))) _Float16 v16h;
typedef __attribute__((ext_vector_type(8)))  _Float16 v8h;
typedef __attribute__((ext_vector_type(8)))  float    v8f;

#define LIF_THRES 1.0f
#define LIF_DECAY 0.9375f

// Load the per-lane K=32 slice for an A or B WMMA operand (16-bit, 16x16x32).
// Per ISA layout: lane l (row/col = l&15, khalf = l>>4) holds K chunks
// [khalf*8 .. +7] and [16+khalf*8 .. +7]  -> two contiguous 16B loads.
__device__ __forceinline__ v16h load_k32(const _Float16* p) {
    v8h lo = *(const v8h*)p;
    v8h hi = *(const v8h*)(p + 16);
    v16h r;
#pragma unroll
    for (int i = 0; i < 8; ++i) { r[i] = lo[i]; r[8 + i] = hi[i]; }
    return r;
}

// ---------------------------------------------------------------------------
// Input integrate-fire (no decay): m += x; s = m>=1; m -= s.
// ---------------------------------------------------------------------------
__global__ void k_input_lif(const float* __restrict__ img,
                            float* __restrict__ mem,
                            _Float16* __restrict__ spk, int n) {
    int i = blockIdx.x * blockDim.x + threadIdx.x;
    if (i >= n) return;
    float v = mem[i] + img[i];
    bool s = v >= LIF_THRES;
    v -= s ? LIF_THRES : 0.0f;
    mem[i] = v;
    spk[i] = s ? (_Float16)1.0f : (_Float16)0.0f;
}

// ---------------------------------------------------------------------------
// im2col for 5x5 VALID conv, square output. dst is (M x Kpad) fp16,
// M = 32*Hout*Hout, k = ic*25 + ky*5 + kx, zero fill for k >= K.
// grid: (ceil(Kpad/128), M), block 128.
// ---------------------------------------------------------------------------
__global__ void k_im2col5(const _Float16* __restrict__ src,
                          _Float16* __restrict__ dst,
                          int C, int Hin, int Win, int Hout, int K, int Kpad) {
    int k = blockIdx.x * blockDim.x + threadIdx.x;
    if (k >= Kpad) return;
    int m = blockIdx.y;
    _Float16 v = (_Float16)0.0f;
    if (k < K) {
        int hw = Hout * Hout;
        int b  = m / hw;
        int rm = m - b * hw;
        int oy = rm / Hout;
        int ox = rm - oy * Hout;
        int ic = k / 25;
        int r  = k - ic * 25;
        int ky = r / 5;
        int kx = r - ky * 5;
        v = src[(((size_t)b * C + ic) * Hin + (oy + ky)) * Win + (ox + kx)];
    }
    dst[(size_t)m * Kpad + k] = v;
}

// ---------------------------------------------------------------------------
// fp32 -> fp16 weight conversion with K (and N) zero padding.
// dst[(n)*Kpad + k] = (n < Nsrc && k < Ksrc) ? src[n*Ksrc + k] : 0
// ---------------------------------------------------------------------------
__global__ void k_wcvt(const float* __restrict__ src, _Float16* __restrict__ dst,
                       int Nsrc, int Ksrc, int Kpad, int total) {
    int e = blockIdx.x * blockDim.x + threadIdx.x;
    if (e >= total) return;
    int n = e / Kpad;
    int k = e - n * Kpad;
    float v = (n < Nsrc && k < Ksrc) ? src[(size_t)n * Ksrc + k] : 0.0f;
    dst[e] = (_Float16)v;
}

// ---------------------------------------------------------------------------
// Fused GEMM + LIF.  C[M,N] = A[M,K] * W[N,K]^T, then LIF on membrane (f32,
// M x N row-major), spikes written fp16 either linear (mode 0, ld = N) or
// scattered to NCHW (mode 1: m = (b,oy,ox), n = channel).
// Block = 128 threads = 4 waves arranged WM x WN; each wave owns a 16(M) x
// 64(N) tile: one A fetch feeds 4 WMMAs.  grid = (M/(16*WM), N/(64*WN)).
// ---------------------------------------------------------------------------
__global__ void k_gemm_lif(const _Float16* __restrict__ A,
                           const _Float16* __restrict__ W,
                           float* __restrict__ mem,
                           _Float16* __restrict__ spk,
                           int M, int N, int K, int lda, int ldw,
                           int WM, int WN,
                           int mode, int Cch, int Hh, int Wd) {
    int lane  = threadIdx.x & 31;
    int wave  = threadIdx.x >> 5;
    int wn    = wave % WN;
    int wm    = wave / WN;
    int m0    = blockIdx.x * (WM * 16) + wm * 16;
    int n0    = blockIdx.y * (WN * 64) + wn * 64;
    int l15   = lane & 15;
    int khalf = lane >> 4;

    const _Float16* ap = A + (size_t)(m0 + l15) * lda + khalf * 8;
    const _Float16* wp = W + (size_t)(n0 + l15) * ldw + khalf * 8;
    size_t wstep = (size_t)16 * ldw;

    v8f acc[4];
#pragma unroll
    for (int j = 0; j < 4; ++j) acc[j] = (v8f)(0.0f);

    for (int k0 = 0; k0 < K; k0 += 32) {
        v16h a = load_k32(ap);
        __builtin_prefetch(ap + 256, 0, 1);          // global_prefetch_b8
        ap += 32;
#pragma unroll
        for (int j = 0; j < 4; ++j) {
            v16h b = load_k32(wp + (size_t)j * wstep);
            acc[j] = __builtin_amdgcn_wmma_f32_16x16x32_f16(
                false, a, false, b, (short)0, acc[j], false, false);
        }
        __builtin_prefetch(wp + 256, 0, 1);
        wp += 32;
    }

    // LIF epilogue. D layout: lane l, vgpr r -> (m = r + 8*(l>>4), n = l&15).
#pragma unroll
    for (int j = 0; j < 4; ++j) {
        int n = n0 + j * 16 + l15;
#pragma unroll
        for (int r = 0; r < 8; ++r) {
            int m = m0 + r + khalf * 8;
            size_t midx = (size_t)m * N + n;
            float v = mem[midx] + acc[j][r];
            bool s = v >= LIF_THRES;
            v = s ? (v - LIF_THRES) : v * LIF_DECAY;
            mem[midx] = v;
            _Float16 sv = s ? (_Float16)1.0f : (_Float16)0.0f;
            if (mode == 0) {
                spk[(size_t)m * N + n] = sv;
            } else {
                int hw = Hh * Wd;
                int b  = m / hw;
                int rm = m - b * hw;
                int oy = rm / Wd;
                int ox = rm - oy * Wd;
                spk[(((size_t)b * Cch + n) * Hh + oy) * Wd + ox] = sv;
            }
        }
    }
}

// ---------------------------------------------------------------------------
// lin2 + LIF: [32,16] = s_l1_prev[32,1024] * Wl2pad[16,1024]^T, spikes for
// n < 10 written to out[b][t+1][n] as float {0,1}.  1 block, 2 waves.
// ---------------------------------------------------------------------------
__global__ void k_lin2_lif(const _Float16* __restrict__ A,
                           const _Float16* __restrict__ W,
                           float* __restrict__ mem,
                           float* __restrict__ out, int t) {
    int lane  = threadIdx.x & 31;
    int wave  = threadIdx.x >> 5;
    int m0    = wave * 16;
    int l15   = lane & 15;
    int khalf = lane >> 4;

    const _Float16* ap = A + (size_t)(m0 + l15) * 1024 + khalf * 8;
    const _Float16* wp = W + (size_t)l15 * 1024 + khalf * 8;

    v8f acc = (v8f)(0.0f);
    for (int k0 = 0; k0 < 1024; k0 += 32) {
        v16h a = load_k32(ap); ap += 32;
        v16h b = load_k32(wp); wp += 32;
        acc = __builtin_amdgcn_wmma_f32_16x16x32_f16(
            false, a, false, b, (short)0, acc, false, false);
    }
#pragma unroll
    for (int r = 0; r < 8; ++r) {
        int m = m0 + r + khalf * 8;     // batch index
        int n = l15;                    // class index (10..15 are padding)
        size_t midx = (size_t)m * 16 + n;
        float v = mem[midx] + acc[r];
        bool s = v >= LIF_THRES;
        v = s ? (v - LIF_THRES) : v * LIF_DECAY;
        mem[midx] = v;
        if (n < 10)
            out[((size_t)m * 21 + (t + 1)) * 10 + n] = s ? 1.0f : 0.0f;
    }
}

// ---------------------------------------------------------------------------
extern "C" void kernel_launch(void* const* d_in, const int* in_sizes, int n_in,
                              void* d_out, int out_size, void* d_ws, size_t ws_size,
                              hipStream_t stream) {
    (void)in_sizes; (void)n_in; (void)out_size; (void)ws_size;
    const float* image = (const float*)d_in[0];   // (32,3,32,32)
    const float* Wc1   = (const float*)d_in[1];   // (64,3,5,5)
    const float* Wc2   = (const float*)d_in[2];   // (128,64,5,5)
    const float* Wl1   = (const float*)d_in[3];   // (1024,73728)
    const float* Wl2   = (const float*)d_in[4];   // (10,1024)
    float* outp = (float*)d_out;                  // (32,21,10) as float {0,1}

    uint8_t* base = (uint8_t*)d_ws;
    size_t off = 0;
    auto carve = [&](size_t bytes) -> void* {
        void* p = base + off;
        off += (bytes + 255) & ~(size_t)255;
        return p;
    };

    // fp16 weights (K/N padded)
    _Float16* Wc1h = (_Float16*)carve((size_t)64 * 96 * 2);        // K 75 -> 96
    _Float16* Wc2h = (_Float16*)carve((size_t)128 * 1600 * 2);
    _Float16* Wl1h = (_Float16*)carve((size_t)1024 * 73728 * 2);   // 151 MB (fits L2)
    _Float16* Wl2h = (_Float16*)carve((size_t)16 * 1024 * 2);      // N 10 -> 16
    // spike tensors (fp16)
    _Float16* s_in  = (_Float16*)carve((size_t)98304 * 2);
    _Float16* A1    = (_Float16*)carve((size_t)25088 * 96 * 2);
    _Float16* A2    = (_Float16*)carve((size_t)18432 * 1600 * 2);
    _Float16* s_c1[2], * s_c2[2], * s_l1[2];
    for (int i = 0; i < 2; ++i) s_c1[i] = (_Float16*)carve((size_t)32 * 64 * 28 * 28 * 2);
    for (int i = 0; i < 2; ++i) s_c2[i] = (_Float16*)carve((size_t)32 * 128 * 24 * 24 * 2);
    for (int i = 0; i < 2; ++i) s_l1[i] = (_Float16*)carve((size_t)32 * 1024 * 2);
    // membranes (f32, GEMM M x N layout)
    float* mem_in = (float*)carve((size_t)98304 * 4);
    float* mem_c1 = (float*)carve((size_t)25088 * 64 * 4);
    float* mem_c2 = (float*)carve((size_t)18432 * 128 * 4);
    float* mem_l1 = (float*)carve((size_t)32 * 1024 * 4);
    float* mem_l2 = (float*)carve((size_t)32 * 16 * 4);

    // Zero all persistent state + output (poisoned by harness; must be
    // deterministic per call).
    hipMemsetAsync(mem_in, 0, (size_t)98304 * 4, stream);
    hipMemsetAsync(mem_c1, 0, (size_t)25088 * 64 * 4, stream);
    hipMemsetAsync(mem_c2, 0, (size_t)18432 * 128 * 4, stream);
    hipMemsetAsync(mem_l1, 0, (size_t)32 * 1024 * 4, stream);
    hipMemsetAsync(mem_l2, 0, (size_t)32 * 16 * 4, stream);
    for (int i = 0; i < 2; ++i) {
        hipMemsetAsync(s_c1[i], 0, (size_t)32 * 64 * 28 * 28 * 2, stream);
        hipMemsetAsync(s_c2[i], 0, (size_t)32 * 128 * 24 * 24 * 2, stream);
        hipMemsetAsync(s_l1[i], 0, (size_t)32 * 1024 * 2, stream);
    }
    hipMemsetAsync(outp, 0, (size_t)32 * 21 * 10 * 4, stream);

    // Weight conversion (once per launch).
    k_wcvt<<<(64 * 96 + 255) / 256, 256, 0, stream>>>(Wc1, Wc1h, 64, 75, 96, 64 * 96);
    k_wcvt<<<(128 * 1600 + 255) / 256, 256, 0, stream>>>(Wc2, Wc2h, 128, 1600, 1600, 128 * 1600);
    k_wcvt<<<(int)(((size_t)1024 * 73728 + 255) / 256), 256, 0, stream>>>(
        Wl1, Wl1h, 1024, 73728, 73728, 1024 * 73728);
    k_wcvt<<<(16 * 1024 + 255) / 256, 256, 0, stream>>>(Wl2, Wl2h, 10, 1024, 1024, 16 * 1024);

    for (int t = 0; t < 20; ++t) {
        int cur = t & 1, prev = cur ^ 1;

        // input IF (spikes consumed same step by conv1)
        k_input_lif<<<(98304 + 255) / 256, 256, 0, stream>>>(image, mem_in, s_in, 98304);

        // conv1: M=25088, N=64, K=75->96
        k_im2col5<<<dim3(1, 25088), 128, 0, stream>>>(s_in, A1, 3, 32, 32, 28, 75, 96);
        k_gemm_lif<<<dim3(392, 1), 128, 0, stream>>>(
            A1, Wc1h, mem_c1, s_c1[cur], 25088, 64, 96, 96, 96,
            /*WM=*/4, /*WN=*/1, /*mode=*/1, 64, 28, 28);

        // conv2 reads previous-step conv1 spikes: M=18432, N=128, K=1600
        k_im2col5<<<dim3(13, 18432), 128, 0, stream>>>(s_c1[prev], A2, 64, 28, 28, 24, 1600, 1600);
        k_gemm_lif<<<dim3(576, 1), 128, 0, stream>>>(
            A2, Wc2h, mem_c2, s_c2[cur], 18432, 128, 1600, 1600, 1600,
            /*WM=*/2, /*WN=*/2, /*mode=*/1, 128, 24, 24);

        // lin1 reads previous-step conv2 spikes (flattened NCHW = 32 x 73728)
        k_gemm_lif<<<dim3(1, 8), 128, 0, stream>>>(
            s_c2[prev], Wl1h, mem_l1, s_l1[cur], 32, 1024, 73728, 73728, 73728,
            /*WM=*/2, /*WN=*/2, /*mode=*/0, 0, 0, 0);

        // lin2 reads previous-step lin1 spikes; writes out[:, t+1, :]
        k_lin2_lif<<<1, 64, 0, stream>>>(s_l1[prev], Wl2h, mem_l2, outp, t);
    }
}
